// SimpleWaveNet_21509196218461
// MI455X (gfx1250) — matrix operand order repaired and checked by reading, not compiled, not run on hardware
//
#include <hip/hip_runtime.h>

// ---------------- types ----------------
typedef __bf16 bf16_t;
typedef __attribute__((ext_vector_type(16))) __bf16 bf16x16;
typedef __attribute__((ext_vector_type(8)))  __bf16 bf16x8;
typedef __attribute__((ext_vector_type(8)))  float  floatx8;

static constexpr int  H       = 1028;
static constexpr int  HP      = 1056;         // K padded to multiple of 32 (and 16B-aligned rows)
static constexpr int  G4      = 4 * H;        // 4112
static constexpr int  NUMAX   = 16;           // hidden units per persistent block
static constexpr int  THREADS = 256;
static constexpr int  KSLICE  = HP / 4;       // 264 floats per K-slice (16B aligned)

// dynamic LDS layout (floats)
static constexpr int W_S_FLOATS  = 4 * NUMAX * HP;          // 67584 (rows padded to HP)
static constexpr int H_S_FLOATS  = HP;                      // 1056
static constexpr int RED_FLOATS  = 256;
static constexpr int RED2_FLOATS = 16;
static constexpr int SMEM_BYTES =
    (W_S_FLOATS + H_S_FLOATS + RED_FLOATS + RED2_FLOATS) * 4;   // 275648 B < 320KB/WGP

// ---------------- grid-wide generation barrier ----------------
__device__ __forceinline__ void grid_barrier(unsigned* cnt, unsigned* gen, unsigned nb) {
    __syncthreads();
    if (threadIdx.x == 0) {
        __threadfence();
        unsigned g = __hip_atomic_load(gen, __ATOMIC_ACQUIRE, __HIP_MEMORY_SCOPE_AGENT);
        unsigned arrived =
            __hip_atomic_fetch_add(cnt, 1u, __ATOMIC_ACQ_REL, __HIP_MEMORY_SCOPE_AGENT);
        if (arrived == nb - 1u) {
            __hip_atomic_store(cnt, 0u, __ATOMIC_RELAXED, __HIP_MEMORY_SCOPE_AGENT);
            __hip_atomic_fetch_add(gen, 1u, __ATOMIC_ACQ_REL, __HIP_MEMORY_SCOPE_AGENT);
        } else {
            while (__hip_atomic_load(gen, __ATOMIC_ACQUIRE, __HIP_MEMORY_SCOPE_AGENT) == g)
                __builtin_amdgcn_s_sleep(2);
        }
    }
    __syncthreads();
}

// ---------------- init: barrier words, out[t]=lin_b, zero h1bf pad columns ----------------
__global__ void init_kernel(unsigned* bar, float* out, const float* lin_b,
                            bf16_t* h1bf, int T) {
    int i = blockIdx.x * blockDim.x + threadIdx.x;
    if (i == 0) { bar[0] = 0u; bar[1] = 0u; }
    if (i < T) out[i] = lin_b[0];
    // zero the K-pad columns of the padded h1 buffer once per call
    const int padW = HP - H;                              // 28
    for (int idx = i; idx < T * padW; idx += gridDim.x * blockDim.x) {
        int t = idx / padW, k = H + (idx - t * padW);
        h1bf[(size_t)t * HP + k] = (bf16_t)0.0f;
    }
}

// ---------------- fp32 -> bf16 weight conversion into padded [G4][HP] ----------------
__global__ void cvt_bf16_pad_kernel(const float* __restrict__ src, bf16_t* __restrict__ dst) {
    int i = blockIdx.x * blockDim.x + threadIdx.x;
    int n = G4 * HP;
    if (i >= n) return;
    int r = i / HP, k = i - r * HP;
    dst[i] = (k < H) ? (bf16_t)src[(size_t)r * H + k] : (bf16_t)0.0f;
}

// ---------------- persistent LSTM layer (mode 0: layer0, mode 1: layer1+head) ----------------
__global__ __launch_bounds__(THREADS, 1)
void lstm_persistent(int mode,
                     const float* __restrict__ xin,    // mode0: x[T]; mode1: xg1[T,4H]
                     const float* __restrict__ w_ih,   // mode0: w_ih0[4H] (in-dim 1)
                     const float* __restrict__ w_hh,   // [4H, H]
                     bf16_t* __restrict__ h1bf,        // mode0 output: h1 bf16 [T, HP] (padded)
                     const float* __restrict__ lin_w,  // mode1: [H]
                     float* __restrict__ out,          // mode1: [T] (pre-init to lin_b)
                     const float* __restrict__ b_ih,
                     const float* __restrict__ b_hh,
                     float* __restrict__ hping,        // [2, H] fp32 ping-pong
                     unsigned* __restrict__ bar,
                     int T, int nBlocks)
{
    extern __shared__ char smem[];
    float* w_s  = (float*)smem;                                       // [nRows][HP]
    float* h_s  = (float*)(smem + W_S_FLOATS * 4);                    // [HP]
    float* red  = (float*)(smem + (W_S_FLOATS + H_S_FLOATS) * 4);     // [256]
    float* red2 = (float*)(smem + (W_S_FLOATS + H_S_FLOATS + RED_FLOATS) * 4); // [16]

    const int tid = threadIdx.x;
    const int blk = blockIdx.x;
    const int u0  = blk * NUMAX;
    const int NU  = (H - u0 < NUMAX) ? (H - u0) : NUMAX;
    const int nRows = 4 * NU;

    // ---- stage recurrent weight slice into LDS once, rows padded to HP with zeros ----
    for (int idx = tid; idx < nRows * HP; idx += THREADS) {
        int lr = idx / HP, k = idx - lr * HP;
        int g = lr / NU, uu = lr - g * NU;
        w_s[idx] = (k < H) ? w_hh[(size_t)(g * H + u0 + uu) * H + k] : 0.0f;
    }

    // per-row constants for gate rows owned by threads tid < nRows
    float bias_r = 0.f, wih_r = 0.f; int grow = 0;
    if (tid < nRows) {
        int g = tid / NU, uu = tid - g * NU;
        grow = g * H + u0 + uu;
        bias_r = b_ih[grow] + b_hh[grow];
        if (mode == 0) wih_r = w_ih[grow];
    }
    __syncthreads();

    const int rid   = tid & 63;    // gate-row id
    const int slice = tid >> 6;    // 4 K-slices of KSLICE floats (16B aligned)
    const int k0    = slice * KSLICE;

    float c_u = 0.f;               // cell state lives in registers for the whole sequence

    for (int t = 0; t < T; ++t) {
        float partial = 0.f;
        if (t > 0) {
            const float* hp = hping + ((t - 1) & 1) * H;
            for (int k = tid; k < HP; k += THREADS) h_s[k] = (k < H) ? hp[k] : 0.0f;
            __syncthreads();
            if (rid < nRows) {
                const float4* wr = (const float4*)(w_s + (size_t)rid * HP + k0);
                const float4* hr = (const float4*)(h_s + k0);
                #pragma unroll 4
                for (int j = 0; j < KSLICE / 4; ++j) {            // 66 ds_load_b128 pairs
                    float4 w4 = wr[j], h4 = hr[j];
                    partial += w4.x * h4.x + w4.y * h4.y + w4.z * h4.z + w4.w * h4.w;
                }
            }
        }
        red[tid] = partial;
        __syncthreads();

        if (tid < nRows) {
            float g = red[tid] + red[tid + 64] + red[tid + 128] + red[tid + 192];
            if (mode == 0) g += wih_r * xin[t] + bias_r;           // xg0 inline (in-dim 1)
            else           g += xin[(size_t)t * G4 + grow];        // precomputed xg1 (biases baked in)
            red[tid] = g;
        }
        __syncthreads();

        if (tid < NU) {
            float iv = 1.f / (1.f + __expf(-red[tid]));
            float fv = 1.f / (1.f + __expf(-red[NU + tid]));
            float gv = tanhf(red[2 * NU + tid]);
            float ov = 1.f / (1.f + __expf(-red[3 * NU + tid]));
            c_u = fv * c_u + iv * gv;
            float h = ov * tanhf(c_u);
            hping[(t & 1) * H + u0 + tid] = h;
            if (mode == 0) h1bf[(size_t)t * HP + u0 + tid] = (bf16_t)h;
            else           red2[tid] = h * lin_w[u0 + tid];
        }

        if (mode == 1) {
            __syncthreads();
            if (tid == 0) {
                float s = 0.f;
                for (int u = 0; u < NU; ++u) s += red2[u];
                atomicAdd(out + t, s);                             // global_atomic_add_f32
            }
            if (tid < nRows && (t + 1) < T)
                __builtin_prefetch(&xin[(size_t)(t + 1) * G4 + grow], 0, 0); // global_prefetch_b8
        }

        __threadfence();
        grid_barrier(bar, bar + 1, (unsigned)nBlocks);
    }
}

// ---------------- layer-1 input GEMM via bf16 WMMA: xg1 = h1 @ w_ih1^T + b ----------------
// Buffers padded to K=HP -> fragment halves are contiguous 16B chunks: 4x global_load_b128/WMMA.
__global__ __launch_bounds__(THREADS)
void xg1_gemm_wmma(const bf16_t* __restrict__ A,   // h1 bf16, [T, HP] row-major (zero-padded K)
                   const bf16_t* __restrict__ W,   // w_ih1 bf16, [G4, HP] row-major (zero-padded K)
                   const float* __restrict__ b_ih,
                   const float* __restrict__ b_hh,
                   float* __restrict__ xg1,        // [T, G4]
                   int T)
{
    const int wave = threadIdx.x >> 5;
    const int lane = threadIdx.x & 31;
    const int jTiles = G4 / 16;                    // 257
    const long nTiles = (long)(T / 16) * jTiles;
    const long tile = (long)blockIdx.x * 8 + wave;
    if (tile >= nTiles) return;                    // wave-uniform

    const int tT = (int)(tile / jTiles);
    const int jT = (int)(tile % jTiles);
    const int m  = lane & 15;
    const int hi = lane >> 4;                      // lane half

    // A layout (matches B): lanes 0-15 K {0..15}, lanes 16-31 K {16..31}
    const int aK1 = hi * 16;
    const int aK2 = hi * 16 + 8;
    const int bK  = hi * 16;                       // B: 16 contiguous K per lane

    const bf16_t* Arow = A + (size_t)(tT * 16 + m) * HP;   // A[m, k]
    const bf16_t* Bcol = W + (size_t)(jT * 16 + m) * HP;   // B[k, n] = W[n*HP + k]

    union frag16 { bf16x16 v; bf16x8 h[2]; };

    floatx8 acc = {};
    #pragma unroll 3
    for (int kc = 0; kc < HP; kc += 32) {          // 33 WMMAs, no guards
        frag16 a, b;
        a.h[0] = *(const bf16x8*)(Arow + kc + aK1);       // global_load_b128
        a.h[1] = *(const bf16x8*)(Arow + kc + aK2);       // global_load_b128
        b.h[0] = *(const bf16x8*)(Bcol + kc + bK);        // global_load_b128
        b.h[1] = *(const bf16x8*)(Bcol + kc + bK + 8);    // global_load_b128
        acc = __builtin_amdgcn_wmma_f32_16x16x32_bf16(
            /*neg_a=*/false, a.v, /*neg_b=*/false, b.v,
            /*c_mod=*/(short)0, acc, /*reuse_a=*/false, /*reuse_b=*/false);
    }

    const int n = jT * 16 + m;
    const float bias = b_ih[n] + b_hh[n];
    #pragma unroll
    for (int v = 0; v < 8; ++v) {
        int M = v + 8 * hi;                        // D layout: VGPRv -> M = v (+8 for upper lanes)
        xg1[(size_t)(tT * 16 + M) * G4 + n] = acc[v] + bias;
    }
}

// ---------------- host launch ----------------
extern "C" void kernel_launch(void* const* d_in, const int* in_sizes, int n_in,
                              void* d_out, int out_size, void* d_ws, size_t ws_size,
                              hipStream_t stream) {
    const float* x     = (const float*)d_in[0];
    const float* w_ih0 = (const float*)d_in[1];
    const float* w_hh0 = (const float*)d_in[2];
    const float* b_ih0 = (const float*)d_in[3];
    const float* b_hh0 = (const float*)d_in[4];
    const float* w_ih1 = (const float*)d_in[5];
    const float* w_hh1 = (const float*)d_in[6];
    const float* b_ih1 = (const float*)d_in[7];
    const float* b_hh1 = (const float*)d_in[8];
    const float* lin_w = (const float*)d_in[9];
    const float* lin_b = (const float*)d_in[10];
    float* out = (float*)d_out;

    const int T = in_sizes[0];                     // 16384

    // workspace carve-up (256B aligned)
    char* ws = (char*)d_ws;
    unsigned* bar = (unsigned*)ws;                                   // 2 words
    float*  hping = (float*)(ws + 256);                              // 2*H fp32
    float*  xg1   = (float*)(ws + 8704);                             // T*G4 fp32
    size_t  xg1Bytes = (size_t)T * G4 * sizeof(float);
    bf16_t* h1bf   = (bf16_t*)(ws + 8704 + xg1Bytes);                // T*HP bf16 (K padded)
    size_t  h1Bytes = (size_t)T * HP * sizeof(bf16_t);
    bf16_t* wih1bf = (bf16_t*)(ws + 8704 + xg1Bytes + h1Bytes);      // G4*HP bf16 (K padded)

    // allow >64KB dynamic LDS for the persistent kernel
    (void)hipFuncSetAttribute(reinterpret_cast<const void*>(lstm_persistent),
                              hipFuncAttributeMaxDynamicSharedMemorySize, SMEM_BYTES);

    init_kernel<<<(T + THREADS - 1) / THREADS, THREADS, 0, stream>>>(bar, out, lin_b, h1bf, T);

    const int nW = G4 * HP;
    cvt_bf16_pad_kernel<<<(nW + THREADS - 1) / THREADS, THREADS, 0, stream>>>(w_ih1, wih1bf);

    const int NB = (H + NUMAX - 1) / NUMAX;        // 65 persistent blocks, 1 per WGP (~269KB LDS each)

    // layer 0: sequential recurrence, weights pinned in LDS, h1 emitted in padded bf16
    lstm_persistent<<<NB, THREADS, SMEM_BYTES, stream>>>(
        0, x, w_ih0, w_hh0, h1bf, nullptr, nullptr, b_ih0, b_hh0, hping, bar, T, NB);

    // layer-1 input precompute: bf16 WMMA GEMM (parallel over all T)
    const long nTiles = (long)(T / 16) * (G4 / 16);
    const int  gemmBlocks = (int)((nTiles + 7) / 8);
    xg1_gemm_wmma<<<gemmBlocks, THREADS, 0, stream>>>(h1bf, wih1bf, b_ih1, b_hh1, xg1, T);

    // layer 1: recurrence + fused 1-wide head (out pre-initialized to lin_b)
    lstm_persistent<<<NB, THREADS, SMEM_BYTES, stream>>>(
        1, xg1, nullptr, w_hh1, nullptr, lin_w, out, b_ih1, b_hh1, hping, bar, T, NB);
}
